// FakeNewsDetectionModel_68444598829542
// MI455X (gfx1250) — compile-verified, matrix operand order checked by reading, not run on hardware
//
#include <hip/hip_runtime.h>
#include <math.h>
#include <stdint.h>

#define S_LEN  4096
#define DMODEL 768
#define NHEAD  12
#define DHEAD  64
#define FDIM   3072
#define NCHUNK 16
#define NBAND  768
#define NEGV   (-1.0e9f)

typedef __attribute__((ext_vector_type(16))) _Float16 v16h;
typedef __attribute__((ext_vector_type(8)))  _Float16 v8h;
typedef __attribute__((ext_vector_type(8)))  float    v8f;
typedef __attribute__((ext_vector_type(4)))  unsigned int v4u;
typedef __attribute__((ext_vector_type(8)))  int v8i;
typedef __attribute__((ext_vector_type(4)))  int v4i;

// ---------------------------------------------------------------------------
// WMMA helpers
// ---------------------------------------------------------------------------
__device__ __forceinline__ v8f wmma16(v16h a, v16h b, v8f c) {
  // D = A(16x32 f16) * B(32x16 f16) + C(16x16 f32)
  return __builtin_amdgcn_wmma_f32_16x16x32_f16(false, a, false, b,
                                                (short)0, c, false, false);
}

// Load a 16x32 (K-contiguous) fragment from LDS.
// lanes 0-15: row M=lane, K[0..7] & K[16..23]; lanes 16-31: K[8..15] & K[24..31].
__device__ __forceinline__ v16h ld_frag(const _Float16* base, int pitch,
                                        int row0, int k0, int lane) {
  int r = lane & 15;
  int koff = k0 + ((lane >> 4) << 3);
  const _Float16* rp = base + (row0 + r) * pitch + koff;
  union { v16h v; v8h h[2]; } u;
  u.h[0] = *(const v8h*)(rp);
  u.h[1] = *(const v8h*)(rp + 16);
  return u.v;
}

// ---------------------------------------------------------------------------
// Tensor Data Mover: async 2D tile (rows x 32 f16 cols) global -> LDS.
// D# per CDNA5 ISA ch.8: group0 {count, lds_addr, global_addr, type=2},
// group1 {data_size=2B, tensor dims, tile dims, dim0 stride}; groups 2/3 = 0.
// This toolchain's builtin takes 6 args (g0, g1, g2, g3, g1b, cpol).
// ---------------------------------------------------------------------------
__device__ __forceinline__ void tdm_load_tile(const _Float16* gptr,
                                              int stride_elems, int rows,
                                              unsigned lds_off) {
  unsigned long long ga = (unsigned long long)(uintptr_t)gptr;
  v4u g0;
  g0[0] = 1u;                                            // count=1, no gather
  g0[1] = lds_off;                                       // LDS byte address
  g0[2] = (unsigned)(ga & 0xFFFFFFFFu);                  // global_addr[31:0]
  g0[3] = (unsigned)((ga >> 32) & 0x01FFFFFFu) | (2u << 30);  // [56:32] | type=2
  v8i g1;
  g1[0] = 1 << 16;                       // data_size = 1 -> 2 bytes/elem
  g1[1] = (int)(32u << 16);              // tensor_dim0 = 32     (bits 79:48)
  g1[2] = (int)((unsigned)rows << 16);   // tensor_dim1 = rows   (bits 111:80)
  g1[3] = (int)(32u << 16);              // tile_dim0 = 32       (bits 127:112)
  g1[4] = rows;                          // tile_dim1 = rows     (bits 143:128)
  g1[5] = stride_elems;                  // tensor_dim0_stride   (bits 207:160)
  g1[6] = 0;
  g1[7] = 0;
  v4i z4 = {};
  v8i z8 = {};
  __builtin_amdgcn_tensor_load_to_lds(g0, g1, z4, z4, z8, 0);
}

__device__ __forceinline__ unsigned lds_offset_of(const void* p) {
  return (unsigned)(uintptr_t)p;  // low 32 bits of generic ptr = LDS offset
}

// ---------------------------------------------------------------------------
// Reductions
// ---------------------------------------------------------------------------
__device__ __forceinline__ float wave_sum(float v) {
#pragma unroll
  for (int o = 16; o > 0; o >>= 1) v += __shfl_xor(v, o, 32);
  return v;
}
__device__ __forceinline__ float wave_max(float v) {
#pragma unroll
  for (int o = 16; o > 0; o >>= 1) v = fmaxf(v, __shfl_xor(v, o, 32));
  return v;
}
__device__ __forceinline__ float block_sum(float v, float* red) {
  int tid = threadIdx.x;
  red[tid] = v; __syncthreads();
  for (int o = 128; o > 0; o >>= 1) {
    if (tid < o) red[tid] += red[tid + o];
    __syncthreads();
  }
  float r = red[0]; __syncthreads();
  return r;
}
__device__ __forceinline__ float block_max(float v, float* red) {
  int tid = threadIdx.x;
  red[tid] = v; __syncthreads();
  for (int o = 128; o > 0; o >>= 1) {
    if (tid < o) red[tid] = fmaxf(red[tid], red[tid + o]);
    __syncthreads();
  }
  float r = red[0]; __syncthreads();
  return r;
}

// ---------------------------------------------------------------------------
// Embedding gather + LayerNorm  (one block per token, 256 threads, D=768)
// ---------------------------------------------------------------------------
__global__ __launch_bounds__(256) void embed_ln_kernel(
    const int* __restrict__ ids, const float* __restrict__ we,
    const float* __restrict__ pe, const float* __restrict__ te,
    const float* __restrict__ g, const float* __restrict__ b,
    float* __restrict__ h, _Float16* __restrict__ h16) {
  __shared__ float red[256];
  int s = blockIdx.x, tid = threadIdx.x;
  int id = ids[s];
  float y[3]; float sum = 0.f;
#pragma unroll
  for (int i = 0; i < 3; ++i) {
    int k = tid + i * 256;
    y[i] = we[(size_t)id * DMODEL + k] + pe[(size_t)(s + 2) * DMODEL + k] + te[k];
    sum += y[i];
  }
  float mu = block_sum(sum, red) * (1.0f / DMODEL);
  float vs = 0.f;
#pragma unroll
  for (int i = 0; i < 3; ++i) { float d = y[i] - mu; vs += d * d; }
  float var = block_sum(vs, red) * (1.0f / DMODEL);
  float rstd = rsqrtf(var + 1e-5f);
#pragma unroll
  for (int i = 0; i < 3; ++i) {
    int k = tid + i * 256;
    float o = (y[i] - mu) * rstd * g[k] + b[k];
    h[(size_t)s * DMODEL + k] = o;
    h16[(size_t)s * DMODEL + k] = (_Float16)o;
  }
}

// ---------------------------------------------------------------------------
// Residual add + LayerNorm
// ---------------------------------------------------------------------------
__global__ __launch_bounds__(256) void add_ln_kernel(
    float* __restrict__ x, const float* __restrict__ a,
    const float* __restrict__ g, const float* __restrict__ b,
    _Float16* __restrict__ x16) {
  __shared__ float red[256];
  int s = blockIdx.x, tid = threadIdx.x;
  float y[3]; float sum = 0.f;
#pragma unroll
  for (int i = 0; i < 3; ++i) {
    int k = tid + i * 256;
    y[i] = x[(size_t)s * DMODEL + k] + a[(size_t)s * DMODEL + k];
    sum += y[i];
  }
  float mu = block_sum(sum, red) * (1.0f / DMODEL);
  float vs = 0.f;
#pragma unroll
  for (int i = 0; i < 3; ++i) { float d = y[i] - mu; vs += d * d; }
  float var = block_sum(vs, red) * (1.0f / DMODEL);
  float rstd = rsqrtf(var + 1e-5f);
#pragma unroll
  for (int i = 0; i < 3; ++i) {
    int k = tid + i * 256;
    float o = (y[i] - mu) * rstd * g[k] + b[k];
    x[(size_t)s * DMODEL + k] = o;
    x16[(size_t)s * DMODEL + k] = (_Float16)o;
  }
}

// ---------------------------------------------------------------------------
// fp32 W[K][N] -> f16 Wt[N][K], 32x32 LDS-tiled so both sides stay coalesced.
// grid (Nd/32, Kd/32), 256 threads.
// ---------------------------------------------------------------------------
__global__ __launch_bounds__(256) void convert_wt_kernel(
    const float* __restrict__ W, _Float16* __restrict__ Wt, int Kd, int Nd) {
  __shared__ _Float16 t[32][33];
  int n0 = blockIdx.x * 32, k0 = blockIdx.y * 32;
  int c = threadIdx.x & 31, r0 = threadIdx.x >> 5;   // 8 row-groups
#pragma unroll
  for (int i = 0; i < 4; ++i) {
    int r = r0 + i * 8;
    t[r][c] = (_Float16)W[(size_t)(k0 + r) * Nd + n0 + c];
  }
  __syncthreads();
#pragma unroll
  for (int i = 0; i < 4; ++i) {
    int r = r0 + i * 8;
    Wt[(size_t)(n0 + r) * Kd + k0 + c] = t[c][r];
  }
}

// ---------------------------------------------------------------------------
// WMMA GEMM: C[M,N] = A[M,K](f16,K-major) * Bt[N,K](f16,K-major)^T + bias
// 128x128 tile, 256 threads (8 waves), each wave 32x64 (2x4 fragments).
// A/B tiles streamed by the Tensor Data Mover with double buffering:
// wave 0 issues two 128x32 tile descriptors per K-step, keeps the next
// pair in flight (s_wait_tensorcnt 2), barrier publishes LDS to all waves.
// act: 0=none, 1=exact GELU. C32/C16 optional.
// ---------------------------------------------------------------------------
__global__ __launch_bounds__(256) void gemm_bias_kernel(
    const _Float16* __restrict__ A, int lda,
    const _Float16* __restrict__ Bt, int ldb,
    const float* __restrict__ bias,
    float* __restrict__ C32, _Float16* __restrict__ C16, int ldc,
    int K, int act) {
  __shared__ _Float16 As[2][128 * 32];
  __shared__ _Float16 Bs[2][128 * 32];
  const int tid = threadIdx.x, lane = tid & 31, wave = tid >> 5;
  const int wm = wave & 3;   // 4 wave-groups along M -> rows wm*32
  const int wn = wave >> 2;  // 2 wave-groups along N -> cols wn*64
  const int blockN = blockIdx.x * 128;
  const int blockM = blockIdx.y * 128;
  const int nsteps = K >> 5;

  const _Float16* Abase = A + (size_t)blockM * lda;
  const _Float16* Bbase = Bt + (size_t)blockN * ldb;
  unsigned ldsA[2] = { lds_offset_of(&As[0][0]), lds_offset_of(&As[1][0]) };
  unsigned ldsB[2] = { lds_offset_of(&Bs[0][0]), lds_offset_of(&Bs[1][0]) };

  v8f zero = {};
  v8f acc[2][4];
#pragma unroll
  for (int i = 0; i < 2; ++i)
#pragma unroll
    for (int j = 0; j < 4; ++j) acc[i][j] = zero;

  if (wave == 0) {  // prologue: stage 0 into buffer 0
    tdm_load_tile(Abase, lda, 128, ldsA[0]);
    tdm_load_tile(Bbase, ldb, 128, ldsB[0]);
  }

  for (int i = 0; i < nsteps; ++i) {
    const int cur = i & 1;
    if (wave == 0) {
      if (i + 1 < nsteps) {
        const int nxt = cur ^ 1;
        tdm_load_tile(Abase + (i + 1) * 32, lda, 128, ldsA[nxt]);
        tdm_load_tile(Bbase + (i + 1) * 32, ldb, 128, ldsB[nxt]);
        __builtin_amdgcn_s_wait_tensorcnt((short)2);  // current pair done
      } else {
        __builtin_amdgcn_s_wait_tensorcnt((short)0);
      }
    }
    __syncthreads();
    v16h af[2], bf[4];
#pragma unroll
    for (int mi = 0; mi < 2; ++mi)
      af[mi] = ld_frag(&As[cur][0], 32, wm * 32 + mi * 16, 0, lane);
#pragma unroll
    for (int ni = 0; ni < 4; ++ni)
      bf[ni] = ld_frag(&Bs[cur][0], 32, wn * 64 + ni * 16, 0, lane);
#pragma unroll
    for (int mi = 0; mi < 2; ++mi)
#pragma unroll
      for (int ni = 0; ni < 4; ++ni) acc[mi][ni] = wmma16(af[mi], bf[ni], acc[mi][ni]);
    __syncthreads();  // all reads done before buffer is re-filled
  }

  const int rhalf = (lane >> 4) << 3;  // 0 or 8
  const int ncol = lane & 15;
#pragma unroll
  for (int mi = 0; mi < 2; ++mi) {
#pragma unroll
    for (int ni = 0; ni < 4; ++ni) {
      int col = blockN + wn * 64 + ni * 16 + ncol;
      float bv = bias ? bias[col] : 0.f;
#pragma unroll
      for (int r = 0; r < 8; ++r) {
        int row = blockM + wm * 32 + mi * 16 + rhalf + r;
        float v = acc[mi][ni][r] + bv;
        if (act == 1) v = 0.5f * v * (1.f + erff(v * 0.70710678118654752f));
        if (C32) C32[(size_t)row * ldc + col] = v;
        if (C16) C16[(size_t)row * ldc + col] = (_Float16)v;
      }
    }
  }
}

// ---------------------------------------------------------------------------
// Band scores: per (h, chunk) S[256,768] = 0.125 * Q[256,64] @ Kband[768,64]^T
// grid: x = n-tile (0..5), y = c*2 + m-tile, z = head.  K rows band-clamped.
// ---------------------------------------------------------------------------
__global__ __launch_bounds__(256) void band_scores_kernel(
    const _Float16* __restrict__ qf, const _Float16* __restrict__ kf,
    _Float16* __restrict__ scores) {
  __shared__ _Float16 As[128 * 64];
  __shared__ _Float16 Bs[128 * 64];
  const int tid = threadIdx.x, lane = tid & 31, wave = tid >> 5;
  const int wm = wave & 3, wn = wave >> 2;
  const int nt = blockIdx.x;
  const int c = blockIdx.y >> 1;
  const int mt = blockIdx.y & 1;
  const int h = blockIdx.z;

#pragma unroll
  for (int i = 0; i < 4; ++i) {
    int idx = tid + i * 256;           // 1024 chunks of 8 -> 128 rows x 8 quads
    int row = idx >> 3;
    int q = (idx & 7) << 3;
    int s = c * 256 + mt * 128 + row;
    *(v8h*)&As[row * 64 + q] = *(const v8h*)(qf + (size_t)s * DMODEL + h * DHEAD + q);
    int j = nt * 128 + row;
    int kp = c * 256 - 256 + j;
    kp = kp < 0 ? 0 : (kp > S_LEN - 1 ? S_LEN - 1 : kp);
    *(v8h*)&Bs[row * 64 + q] = *(const v8h*)(kf + (size_t)kp * DMODEL + h * DHEAD + q);
  }
  __syncthreads();

  v8f zero = {};
  v8f acc[2][4];
#pragma unroll
  for (int i = 0; i < 2; ++i)
#pragma unroll
    for (int j = 0; j < 4; ++j) acc[i][j] = zero;

#pragma unroll
  for (int k0 = 0; k0 < 64; k0 += 32) {
    v16h af[2], bf[4];
#pragma unroll
    for (int mi = 0; mi < 2; ++mi) af[mi] = ld_frag(As, 64, wm * 32 + mi * 16, k0, lane);
#pragma unroll
    for (int ni = 0; ni < 4; ++ni) bf[ni] = ld_frag(Bs, 64, wn * 64 + ni * 16, k0, lane);
#pragma unroll
    for (int mi = 0; mi < 2; ++mi)
#pragma unroll
      for (int ni = 0; ni < 4; ++ni) acc[mi][ni] = wmma16(af[mi], bf[ni], acc[mi][ni]);
  }

  const int rhalf = (lane >> 4) << 3;
  const int ncol = lane & 15;
  _Float16* srow = scores + ((size_t)(h * NCHUNK + c) * 256) * NBAND;
#pragma unroll
  for (int mi = 0; mi < 2; ++mi)
#pragma unroll
    for (int ni = 0; ni < 4; ++ni) {
      int j = nt * 128 + wn * 64 + ni * 16 + ncol;
#pragma unroll
      for (int r = 0; r < 8; ++r) {
        int qi = mt * 128 + wm * 32 + mi * 16 + rhalf + r;
        srow[(size_t)qi * NBAND + j] = (_Float16)(acc[mi][ni][r] * 0.125f);
      }
    }
}

// ---------------------------------------------------------------------------
// Softmax over [glb | 768 band] per row; one wave per row (8 rows/block).
// Computes s_glb = 0.125*q.k0 inline, masks band, writes probs f16 in place
// plus p0 (prob of the global key) as f32.
// ---------------------------------------------------------------------------
__global__ __launch_bounds__(256) void band_softmax_kernel(
    _Float16* __restrict__ scores, const _Float16* __restrict__ qf,
    const _Float16* __restrict__ kf, const int* __restrict__ mask,
    float* __restrict__ p0) {
  const int lane = threadIdx.x & 31, wave = threadIdx.x >> 5;
  int row = blockIdx.x * 8 + wave;          // 0 .. H*NC*256-1
  int h = row / (NCHUNK * 256);
  int rem = row % (NCHUNK * 256);
  int c = rem >> 8;
  int qi = rem & 255;
  int s = c * 256 + qi;

  float dot = 0.f;
#pragma unroll
  for (int u = 0; u < 2; ++u) {
    int d = lane + u * 32;
    dot += (float)qf[(size_t)s * DMODEL + h * DHEAD + d] *
           (float)kf[h * DHEAD + d];
  }
  float sglb = wave_sum(dot) * 0.125f;

  _Float16* srow = scores + ((size_t)(h * NCHUNK + c) * 256 + qi) * NBAND;
  float vals[24];
  float m = sglb;
#pragma unroll
  for (int t = 0; t < 24; ++t) {
    int j = lane + t * 32;
    int kpos = c * 256 - 256 + j;
    bool inr = (kpos >= 0) && (kpos < S_LEN);
    int kpc = kpos < 0 ? 0 : (kpos > S_LEN - 1 ? S_LEN - 1 : kpos);
    int dq = s - kpos; if (dq < 0) dq = -dq;
    bool ok = inr && (dq <= 256) && (kpos != 0) && (mask[kpc] > 0);
    float v = ok ? (float)srow[j] : NEGV;
    vals[t] = v;
    m = fmaxf(m, v);
  }
  m = wave_max(m);
  float sum = 0.f;
#pragma unroll
  for (int t = 0; t < 24; ++t) { vals[t] = __expf(vals[t] - m); sum += vals[t]; }
  sum = wave_sum(sum);
  float e0 = __expf(sglb - m);
  float inv = 1.f / (sum + e0);
#pragma unroll
  for (int t = 0; t < 24; ++t) srow[lane + t * 32] = (_Float16)(vals[t] * inv);
  if (lane == 0) p0[row] = e0 * inv;
}

// ---------------------------------------------------------------------------
// ctx[256,64] = P[256,768] @ Vband[768,64] + p0*v0 per (chunk, head).
// grid(NC, H); 8 waves, each 32 rows x 64 cols; V transposed through LDS.
// ---------------------------------------------------------------------------
__global__ __launch_bounds__(256) void pv_kernel(
    const _Float16* __restrict__ probs, const _Float16* __restrict__ vf,
    const float* __restrict__ p0, _Float16* __restrict__ ctx) {
  __shared__ _Float16 As[256 * 32];
  __shared__ _Float16 Bs[64 * 40];
  const int tid = threadIdx.x, lane = tid & 31, wave = tid >> 5;
  const int c = blockIdx.x, h = blockIdx.y;

  v8f zero = {};
  v8f acc[2][4];
#pragma unroll
  for (int i = 0; i < 2; ++i)
#pragma unroll
    for (int j = 0; j < 4; ++j) acc[i][j] = zero;

  const _Float16* P = probs + (size_t)(h * NCHUNK + c) * 256 * NBAND;

  for (int k0 = 0; k0 < NBAND; k0 += 32) {
#pragma unroll
    for (int i = 0; i < 4; ++i) {
      int idx = tid + i * 256;          // 1024 chunks -> 256 rows x 4 quads
      int row = idx >> 2;
      int q = (idx & 3) << 3;
      *(v8h*)&As[row * 32 + q] = *(const v8h*)(P + (size_t)row * NBAND + k0 + q);
    }
    {
      int jj = tid >> 3;                // 0..31 band rows of this k-step
      int q = (tid & 7) << 3;           // 8-col group of DH
      int kp = c * 256 - 256 + k0 + jj;
      kp = kp < 0 ? 0 : (kp > S_LEN - 1 ? S_LEN - 1 : kp);
      v8h vv = *(const v8h*)(vf + (size_t)kp * DMODEL + h * DHEAD + q);
#pragma unroll
      for (int u = 0; u < 8; ++u) Bs[(q + u) * 40 + jj] = vv[u];  // transpose
    }
    __syncthreads();
    v16h af[2], bf[4];
#pragma unroll
    for (int mi = 0; mi < 2; ++mi) af[mi] = ld_frag(As, 32, wave * 32 + mi * 16, 0, lane);
#pragma unroll
    for (int ni = 0; ni < 4; ++ni) bf[ni] = ld_frag(Bs, 40, ni * 16, 0, lane);
#pragma unroll
    for (int mi = 0; mi < 2; ++mi)
#pragma unroll
      for (int ni = 0; ni < 4; ++ni) acc[mi][ni] = wmma16(af[mi], bf[ni], acc[mi][ni]);
    __syncthreads();
  }

  const int rhalf = (lane >> 4) << 3;
  const int ncol = lane & 15;
#pragma unroll
  for (int mi = 0; mi < 2; ++mi)
#pragma unroll
    for (int ni = 0; ni < 4; ++ni) {
      int d = ni * 16 + ncol;
      float v0 = (float)vf[h * DHEAD + d];   // v at sequence position 0
#pragma unroll
      for (int r = 0; r < 8; ++r) {
        int m = wave * 32 + mi * 16 + rhalf + r;
        int s = c * 256 + m;
        float val = acc[mi][ni][r] + p0[(size_t)(h * NCHUNK + c) * 256 + m] * v0;
        ctx[(size_t)s * DMODEL + h * DHEAD + d] = (_Float16)val;
      }
    }
}

// ---------------------------------------------------------------------------
// qg = x[0] @ qg_w + qg_b   (768 outputs)
// ---------------------------------------------------------------------------
__global__ __launch_bounds__(256) void qg_proj_kernel(
    const float* __restrict__ x0, const float* __restrict__ w,
    const float* __restrict__ b, float* __restrict__ qg) {
  int col = blockIdx.x * 256 + threadIdx.x;
  if (col < DMODEL) {
    float a = b[col];
    for (int k = 0; k < DMODEL; ++k) a += x0[k] * w[(size_t)k * DMODEL + col];
    qg[col] = a;
  }
}

// ---------------------------------------------------------------------------
// Global-token attention: per head, softmax(0.125*qg.kg over S) . vg -> ctx row 0
// ---------------------------------------------------------------------------
__global__ __launch_bounds__(256) void global_attn_kernel(
    const float* __restrict__ qg, const _Float16* __restrict__ kgf,
    const _Float16* __restrict__ vgf, const int* __restrict__ mask,
    _Float16* __restrict__ ctx) {
  __shared__ float sg[S_LEN];
  __shared__ float red[256];
  int h = blockIdx.x, tid = threadIdx.x;

  for (int s = tid; s < S_LEN; s += 256) {
    float d = 0.f;
    for (int k = 0; k < DHEAD; ++k)
      d += qg[h * DHEAD + k] * (float)kgf[(size_t)s * DMODEL + h * DHEAD + k];
    sg[s] = (mask[s] > 0) ? d * 0.125f : NEGV;
  }
  __syncthreads();
  float m = -3.0e38f;
  for (int s = tid; s < S_LEN; s += 256) m = fmaxf(m, sg[s]);
  m = block_max(m, red);
  float sum = 0.f;
  for (int s = tid; s < S_LEN; s += 256) { float e = __expf(sg[s] - m); sg[s] = e; sum += e; }
  sum = block_sum(sum, red);
  float inv = 1.f / sum;
  __syncthreads();
  int d = tid & 63, part = tid >> 6;
  float acc = 0.f;
  for (int s = part; s < S_LEN; s += 4)
    acc += sg[s] * (float)vgf[(size_t)s * DMODEL + h * DHEAD + d];
  red[tid] = acc; __syncthreads();
  if (part == 0) {
    float tot = red[d] + red[64 + d] + red[128 + d] + red[192 + d];
    ctx[h * DHEAD + d] = (_Float16)(tot * inv);  // sequence position 0
  }
}

// ---------------------------------------------------------------------------
// out[2] = h[0,:] @ fc_w + fc_b
// ---------------------------------------------------------------------------
__global__ __launch_bounds__(256) void pooled_fc_kernel(
    const float* __restrict__ h, const float* __restrict__ w,
    const float* __restrict__ b, float* __restrict__ out) {
  __shared__ float red[256];
  int tid = threadIdx.x;
  float p0 = 0.f, p1 = 0.f;
  for (int k = tid; k < DMODEL; k += 256) {
    float x = h[k];
    p0 += x * w[k * 2 + 0];
    p1 += x * w[k * 2 + 1];
  }
  float t0 = block_sum(p0, red);
  float t1 = block_sum(p1, red);
  if (tid == 0) { out[0] = t0 + b[0]; out[1] = t1 + b[1]; }
}

// ---------------------------------------------------------------------------
// Host orchestration
// ---------------------------------------------------------------------------
extern "C" void kernel_launch(void* const* d_in, const int* in_sizes, int n_in,
                              void* d_out, int out_size, void* d_ws, size_t ws_size,
                              hipStream_t stream) {
  (void)in_sizes; (void)n_in; (void)out_size; (void)ws_size;
  const int* ids      = (const int*)d_in[0];
  const int* mask     = (const int*)d_in[1];
  const float* word_e = (const float*)d_in[2];
  const float* pos_e  = (const float*)d_in[3];
  const float* type_e = (const float*)d_in[4];
  const float* emb_g  = (const float*)d_in[5];
  const float* emb_b  = (const float*)d_in[6];
  const float* fc_w   = (const float*)d_in[7];
  const float* fc_b   = (const float*)d_in[8];

  // workspace bump allocator (256B aligned)
  char* wp = (char*)d_ws;
  auto alloc = [&](size_t n) { char* r = wp; wp += (n + 255) & ~(size_t)255; return r; };
  float*    h32    = (float*)   alloc((size_t)S_LEN * DMODEL * 4);
  _Float16* h16    = (_Float16*)alloc((size_t)S_LEN * DMODEL * 2);
  _Float16* q16    = (_Float16*)alloc((size_t)S_LEN * DMODEL * 2);
  _Float16* k16    = (_Float16*)alloc((size_t)S_LEN * DMODEL * 2);
  _Float16* v16    = (_Float16*)alloc((size_t)S_LEN * DMODEL * 2);
  _Float16* kg16   = (_Float16*)alloc((size_t)S_LEN * DMODEL * 2);
  _Float16* vg16   = (_Float16*)alloc((size_t)S_LEN * DMODEL * 2);
  _Float16* ctx16  = (_Float16*)alloc((size_t)S_LEN * DMODEL * 2);
  float*    tmp32  = (float*)   alloc((size_t)S_LEN * DMODEL * 4);
  _Float16* mid16  = (_Float16*)alloc((size_t)S_LEN * FDIM * 2);
  _Float16* sc16   = (_Float16*)alloc((size_t)NHEAD * NCHUNK * 256 * NBAND * 2);
  float*    p0buf  = (float*)   alloc((size_t)NHEAD * NCHUNK * 256 * 4);
  float*    qgbuf  = (float*)   alloc((size_t)DMODEL * 4);
  _Float16* wT     = (_Float16*)alloc((size_t)DMODEL * FDIM * 2);

  const dim3 blk(256);
  const dim3 gDD(DMODEL / 128, S_LEN / 128);   // C: S x D
  const dim3 gDF(FDIM / 128, S_LEN / 128);     // C: S x F
  const dim3 cvDD(DMODEL / 32, DMODEL / 32);   // W: D x D
  const dim3 cvDF(FDIM / 32, DMODEL / 32);     // W: D x F
  const dim3 cvFD(DMODEL / 32, FDIM / 32);     // W: F x D

  embed_ln_kernel<<<S_LEN, blk, 0, stream>>>(ids, word_e, pos_e, type_e,
                                             emb_g, emb_b, h32, h16);

  for (int l = 0; l < 2; ++l) {
    const int base = 9 + l * 22;
    const float* q_w  = (const float*)d_in[base + 0];
    const float* q_b  = (const float*)d_in[base + 1];
    const float* k_w  = (const float*)d_in[base + 2];
    const float* k_b  = (const float*)d_in[base + 3];
    const float* v_w  = (const float*)d_in[base + 4];
    const float* v_b  = (const float*)d_in[base + 5];
    const float* qg_w = (const float*)d_in[base + 6];
    const float* qg_b = (const float*)d_in[base + 7];
    const float* kg_w = (const float*)d_in[base + 8];
    const float* kg_b = (const float*)d_in[base + 9];
    const float* vg_w = (const float*)d_in[base + 10];
    const float* vg_b = (const float*)d_in[base + 11];
    const float* o_w  = (const float*)d_in[base + 12];
    const float* o_b  = (const float*)d_in[base + 13];
    const float* ln1g = (const float*)d_in[base + 14];
    const float* ln1b = (const float*)d_in[base + 15];
    const float* f1_w = (const float*)d_in[base + 16];
    const float* f1_b = (const float*)d_in[base + 17];
    const float* f2_w = (const float*)d_in[base + 18];
    const float* f2_b = (const float*)d_in[base + 19];
    const float* ln2g = (const float*)d_in[base + 20];
    const float* ln2b = (const float*)d_in[base + 21];

    // ---- projections (f16 outputs) ----
    convert_wt_kernel<<<cvDD, blk, 0, stream>>>(q_w, wT, DMODEL, DMODEL);
    gemm_bias_kernel<<<gDD, blk, 0, stream>>>(h16, DMODEL, wT, DMODEL, q_b,
                                              nullptr, q16, DMODEL, DMODEL, 0);
    convert_wt_kernel<<<cvDD, blk, 0, stream>>>(k_w, wT, DMODEL, DMODEL);
    gemm_bias_kernel<<<gDD, blk, 0, stream>>>(h16, DMODEL, wT, DMODEL, k_b,
                                              nullptr, k16, DMODEL, DMODEL, 0);
    convert_wt_kernel<<<cvDD, blk, 0, stream>>>(v_w, wT, DMODEL, DMODEL);
    gemm_bias_kernel<<<gDD, blk, 0, stream>>>(h16, DMODEL, wT, DMODEL, v_b,
                                              nullptr, v16, DMODEL, DMODEL, 0);
    convert_wt_kernel<<<cvDD, blk, 0, stream>>>(kg_w, wT, DMODEL, DMODEL);
    gemm_bias_kernel<<<gDD, blk, 0, stream>>>(h16, DMODEL, wT, DMODEL, kg_b,
                                              nullptr, kg16, DMODEL, DMODEL, 0);
    convert_wt_kernel<<<cvDD, blk, 0, stream>>>(vg_w, wT, DMODEL, DMODEL);
    gemm_bias_kernel<<<gDD, blk, 0, stream>>>(h16, DMODEL, wT, DMODEL, vg_b,
                                              nullptr, vg16, DMODEL, DMODEL, 0);

    // ---- band attention ----
    band_scores_kernel<<<dim3(NBAND / 128, 2 * NCHUNK, NHEAD), blk, 0, stream>>>(
        q16, k16, sc16);
    band_softmax_kernel<<<NHEAD * NCHUNK * 256 / 8, blk, 0, stream>>>(
        sc16, q16, k16, mask, p0buf);
    pv_kernel<<<dim3(NCHUNK, NHEAD), blk, 0, stream>>>(sc16, v16, p0buf, ctx16);

    // ---- global-token attention (overwrites ctx row 0) ----
    qg_proj_kernel<<<(DMODEL + 255) / 256, blk, 0, stream>>>(h32, qg_w, qg_b, qgbuf);
    global_attn_kernel<<<NHEAD, blk, 0, stream>>>(qgbuf, kg16, vg16, mask, ctx16);

    // ---- output projection + residual LN ----
    convert_wt_kernel<<<cvDD, blk, 0, stream>>>(o_w, wT, DMODEL, DMODEL);
    gemm_bias_kernel<<<gDD, blk, 0, stream>>>(ctx16, DMODEL, wT, DMODEL, o_b,
                                              tmp32, nullptr, DMODEL, DMODEL, 0);
    add_ln_kernel<<<S_LEN, blk, 0, stream>>>(h32, tmp32, ln1g, ln1b, h16);

    // ---- FFN ----
    convert_wt_kernel<<<cvDF, blk, 0, stream>>>(f1_w, wT, DMODEL, FDIM);
    gemm_bias_kernel<<<gDF, blk, 0, stream>>>(h16, DMODEL, wT, DMODEL, f1_b,
                                              nullptr, mid16, FDIM, DMODEL, 1);
    convert_wt_kernel<<<cvFD, blk, 0, stream>>>(f2_w, wT, FDIM, DMODEL);
    gemm_bias_kernel<<<gDD, blk, 0, stream>>>(mid16, FDIM, wT, FDIM, f2_b,
                                              tmp32, nullptr, DMODEL, FDIM, 0);
    add_ln_kernel<<<S_LEN, blk, 0, stream>>>(h32, tmp32, ln2g, ln2b, h16);
  }

  pooled_fc_kernel<<<1, blk, 0, stream>>>(h32, fc_w, fc_b, (float*)d_out);
}
